// GraphAttentionLayer_24498493456637
// MI455X (gfx1250) — matrix operand order repaired and checked by reading, not compiled, not run on hardware
//
#include <hip/hip_runtime.h>
#include <cmath>

#define B_     8
#define N_     2048
#define FIN    256
#define FOUT   128
#define ALPHA  0.2f
#define CUTOFF 0.1f
#define NEGINF -9e15f

typedef __attribute__((ext_vector_type(16))) _Float16 v16h;
typedef __attribute__((ext_vector_type(8)))  float    v8f;

union H8 { uint4 q; _Float16 h[8]; };

// load 16 halfs as two b128 loads from two (possibly non-adjacent) 16B-aligned spots
__device__ __forceinline__ v16h load16h(const _Float16* p0, const _Float16* p1) {
    H8 u0, u1;
    u0.q = *(const uint4*)p0;
    u1.q = *(const uint4*)p1;
    v16h v;
#pragma unroll
    for (int i = 0; i < 8; ++i) { v[i] = u0.h[i]; v[8 + i] = u1.h[i]; }
    return v;
}

// ---------------------------------------------------------------- converts
__global__ void cvt_f32_f16(const float* __restrict__ in, _Float16* __restrict__ out, int n) {
    int i = blockIdx.x * blockDim.x + threadIdx.x;
    if (i < n) out[i] = (_Float16)in[i];
}

// W [256,128] row-major f32 -> WT [128,256] f16 (transposed: B-operand friendly)
__global__ void cvt_W_T(const float* __restrict__ in, _Float16* __restrict__ out) {
    int i = blockIdx.x * blockDim.x + threadIdx.x;
    if (i >= FIN * FOUT) return;
    int k = i >> 7, n = i & 127;
    out[n * FIN + k] = (_Float16)in[i];
}

// ---------------------------------------------------------------- hT = (x @ W)^T  per batch: hT[b][n][row] f16
// grid: B*(N/16)=1024 WGs, 256 thr = 8 waves; wave w -> features w*16..w*16+15 of a 16-row tile.
__global__ __launch_bounds__(256) void gemm_h(const _Float16* __restrict__ xf,
                                              const _Float16* __restrict__ WT,
                                              _Float16* __restrict__ hT) {
    const int tid  = threadIdx.x;
    const int wg   = blockIdx.x;
    const int b    = wg >> 7;
    const int row0 = (wg & 127) * 16;
    const int wave = tid >> 5, lane = tid & 31;
    const int hi   = lane >> 4, ln = lane & 15;
    const int nb   = wave * 16 + ln;

    // A: lane holds row (row0+ln); halfs j -> K=k0+16*hi+j
    const _Float16* arow = xf + ((size_t)b * N_ + row0 + ln) * FIN + 16 * hi;
    // B: lane holds column nb; halfs j -> K=k0+16*hi+j  (contiguous 32B in WT)
    const _Float16* brow = WT + (size_t)nb * FIN + 16 * hi;

    v8f c = {};
#pragma unroll
    for (int k0 = 0; k0 < FIN; k0 += 32) {
        v16h a  = load16h(arow + k0, arow + k0 + 8);
        v16h bm = load16h(brow + k0, brow + k0 + 8);
        c = __builtin_amdgcn_wmma_f32_16x16x32_f16(false, a, false, bm, (short)0, c, false, false);
    }
    // C: lane (hi,ln), VGPR r -> row m=r+8*hi, col nb. Rows contiguous in hT -> one b128 store.
    H8 pk;
#pragma unroll
    for (int r = 0; r < 8; ++r) pk.h[r] = (_Float16)c[r];
    *(uint4*)&hT[((size_t)b * FOUT + nb) * N_ + row0 + 8 * hi] = pk.q;
}

// ---------------------------------------------------------------- f_src/f_dst GEMVs (coalesced over rows)
__global__ void fsrcdst(const _Float16* __restrict__ hT, const float* __restrict__ a,
                        float* __restrict__ fs, float* __restrict__ fd) {
    int i = blockIdx.x * blockDim.x + threadIdx.x;
    if (i >= B_ * N_) return;
    const int b = i >> 11, row = i & 2047;
    const _Float16* hb = hT + (size_t)b * FOUT * N_;
    float s0 = 0.f, s1 = 0.f;
    for (int n = 0; n < FOUT; ++n) {
        float h = (float)hb[(size_t)n * N_ + row];   // consecutive threads -> consecutive addresses
        s0 += h * a[n];
        s1 += h * a[FOUT + n];
    }
    fs[i] = s0; fd[i] = s1;
}

// ---------------------------------------------------------------- fused flash attention
// grid: B*(N/16)=1024 WGs, 256 thr = 8 waves. Wave w: WMMA features w*16..; softmax rows 2w, 2w+1.
__global__ __launch_bounds__(256) void attn(const _Float16* __restrict__ hT,
                                            const float* __restrict__ coord,
                                            const float* __restrict__ fs,
                                            const float* __restrict__ fd,
                                            float* __restrict__ out) {
    __shared__ __align__(16) _Float16 PsA[32 * 16];   // P tile pre-swizzled to WMMA A layout
    __shared__ __align__(16) float    scale_s[16];
    __shared__ __align__(16) float    l_s[16];

    const int tid  = threadIdx.x;
    const int wg   = blockIdx.x;
    const int b    = wg >> 7;
    const int row0 = (wg & 127) * 16;
    const int wave = tid >> 5, lane = tid & 31;
    const int hi   = lane >> 4, ln = lane & 15;
    const int nb   = wave * 16 + ln;

    const size_t bn = (size_t)b * N_;
    // softmax row owned by this half-wave (replicated state across its 16 lanes)
    const int   myrow = 2 * wave + hi;
    const float fsr   = fs[bn + row0 + myrow];
    const float rx = coord[(bn + row0 + myrow) * 3 + 0];
    const float ry = coord[(bn + row0 + myrow) * 3 + 1];
    const float rz = coord[(bn + row0 + myrow) * 3 + 2];

    // B operand: lane holds column nb of h; halfs j -> row K=c0+16*hi+j (contiguous in hT)
    const _Float16* bptr = hT + ((size_t)b * FOUT + nb) * N_ + 16 * hi;

    float mrow = NEGINF, lrow = 0.f;
    v8f c = {};

    for (int c0 = 0; c0 < N_; c0 += 32) {
        __syncthreads();                       // prior iter done with PsA/scale_s

        // ---- logits for (myrow, c0+ln) and (myrow, c0+ln+16), straight from global (L2-resident)
        float lg[2];
#pragma unroll
        for (int t = 0; t < 2; ++t) {
            const int col = c0 + ln + 16 * t;
            float ev = fsr + fd[bn + col];
            ev = ev > 0.f ? ev : ALPHA * ev;                 // leaky relu
            const float dx = rx - coord[(bn + col) * 3 + 0];
            const float dy = ry - coord[(bn + col) * 3 + 1];
            const float dz = rz - coord[(bn + col) * 3 + 2];
            const float loc = __expf(-CUTOFF * (dx * dx + dy * dy + dz * dz));
            lg[t] = (loc > 0.01f) ? ev * loc : NEGINF;
        }

        // ---- online softmax: reduce across the 16-lane half-group (xor 1,2,4,8 stays in group)
        float tmax = fmaxf(lg[0], lg[1]);
#pragma unroll
        for (int off = 1; off < 16; off <<= 1) tmax = fmaxf(tmax, __shfl_xor(tmax, off, 32));
        const float mn = fmaxf(mrow, tmax);
        const float sc = __expf(mrow - mn);
        const float p0 = __expf(lg[0] - mn);
        const float p1 = __expf(lg[1] - mn);
        float ts = p0 + p1;
#pragma unroll
        for (int off = 1; off < 16; off <<= 1) ts += __shfl_xor(ts, off, 32);
        lrow = lrow * sc + ts;
        mrow = mn;
        if (ln == 0) scale_s[myrow] = sc;

        // ---- scatter P into WMMA-A layout: slot[(row + 16*hi_k)*16 + j] = P[row][kA(j,hi_k)]
#pragma unroll
        for (int t = 0; t < 2; ++t) {
            const int k   = ln + 16 * t;
            const int hik = (k >> 4) & 1;
            const int j   = k & 15;
            PsA[(myrow + 16 * hik) * 16 + j] = (_Float16)(t ? p1 : p0);
        }
        __syncthreads();                       // PsA + scale_s visible

        // ---- flash rescale of accumulators (two b128 LDS loads)
        const float4 s0 = *(const float4*)&scale_s[8 * hi];
        const float4 s1 = *(const float4*)&scale_s[8 * hi + 4];
        c[0] *= s0.x; c[1] *= s0.y; c[2] *= s0.z; c[3] *= s0.w;
        c[4] *= s1.x; c[5] *= s1.y; c[6] *= s1.z; c[7] *= s1.w;

        // ---- operands: 2x ds_load_b128 (A) + 2x global_load_b128 (B)
        v16h a  = load16h(&PsA[lane * 16], &PsA[lane * 16 + 8]);
        v16h bm = load16h(bptr + c0, bptr + c0 + 8);
        if (c0 + 32 < N_) __builtin_prefetch((const void*)(bptr + c0 + 32), 0, 1);
        c = __builtin_amdgcn_wmma_f32_16x16x32_f16(false, a, false, bm, (short)0, c, false, false);
    }

    if (ln == 0) l_s[myrow] = lrow;
    __syncthreads();

    // ---- finalize: 1/l, ELU, coalesced f32 stores (lane index n is contiguous)
#pragma unroll
    for (int r = 0; r < 8; ++r) {
        const int m = r + 8 * hi;
        float v = c[r] / l_s[m];
        v = v > 0.f ? v : (__expf(v) - 1.f);
        out[(bn + row0 + m) * FOUT + nb] = v;
    }
}

// ---------------------------------------------------------------- launch
extern "C" void kernel_launch(void* const* d_in, const int* in_sizes, int n_in,
                              void* d_out, int out_size, void* d_ws, size_t ws_size,
                              hipStream_t stream) {
    const float* x     = (const float*)d_in[0];   // [8,2048,256]
    const float* coord = (const float*)d_in[1];   // [8,2048,3]
    const float* W     = (const float*)d_in[2];   // [256,128]
    const float* a     = (const float*)d_in[3];   // [256,1]
    float* out = (float*)d_out;                   // [8,2048,128]

    char* ws = (char*)d_ws;
    _Float16* WT = (_Float16*)ws;                              //   64 KB  [128][256]
    _Float16* xf = (_Float16*)(ws + (size_t)65536);            //    8 MB  [8][2048][256]
    _Float16* hT = (_Float16*)(ws + (size_t)65536 + 8388608);  //    4 MB  [8][128][2048]
    float*    fs = (float*)  (ws + (size_t)12648448);          //   64 KB
    float*    fd = (float*)  (ws + (size_t)12713984);          //   64 KB

    const int nW = FIN * FOUT;            // 32768
    const int nX = B_ * N_ * FIN;         // 4194304
    cvt_W_T   <<<(nW + 255) / 256, 256, 0, stream>>>(W, WT);
    cvt_f32_f16<<<(nX + 255) / 256, 256, 0, stream>>>(x, xf, nX);
    gemm_h    <<<B_ * (N_ / 16), 256, 0, stream>>>(xf, WT, hT);
    fsrcdst   <<<(B_ * N_ + 255) / 256, 256, 0, stream>>>(hT, a, fs, fd);
    attn      <<<B_ * (N_ / 16), 256, 0, stream>>>(hT, coord, fs, fd, out);
}
